// PointnetSAModuleMSG_85358180040851
// MI455X (gfx1250) — matrix operand order repaired and checked by reading, not compiled, not run on hardware
//
#include <hip/hip_runtime.h>
#include <hip/hip_bf16.h>

typedef __attribute__((ext_vector_type(16))) _Float16 v16h;
typedef __attribute__((ext_vector_type(8)))  float    v8f;

#define B_ 4
#define N_ 8192
#define C_ 64
#define S_ 2048

// ============================ FPS ============================
// One workgroup per batch (inherently sequential outer loop). wave32 shuffle
// argmax inside each wave, one LDS slot per wave, wave-0 shuffle across waves:
// 3 barriers per iteration. Tie-break -> smallest index (jnp.argmax semantics).
#define FPS_T 1024
#define FPS_PPT (N_ / FPS_T)   // 8

__global__ void __launch_bounds__(FPS_T)
fps_kernel(const float* __restrict__ xyz, float* __restrict__ newxyz) {
  int b = blockIdx.x;
  int t = threadIdx.x;
  int lane = t & 31, wid = t >> 5;        // 32 waves
  const float* px = xyz + (size_t)b * N_ * 3;
  float lx[FPS_PPT], ly[FPS_PPT], lz[FPS_PPT], ld[FPS_PPT];
  #pragma unroll
  for (int i = 0; i < FPS_PPT; ++i) {
    int p = t + i * FPS_T;
    lx[i] = px[p*3+0]; ly[i] = px[p*3+1]; lz[i] = px[p*3+2];
    ld[i] = 3.4e38f;
  }
  __shared__ float sv[32];
  __shared__ int   si[32];
  __shared__ float fpt[3];
  __shared__ int   sfar;
  if (t == 0) sfar = 0;
  __syncthreads();
  for (int j = 0; j < 2048; ++j) {
    if (t == 0) {
      int f = sfar;
      float fx = px[f*3+0], fy = px[f*3+1], fz = px[f*3+2];
      fpt[0] = fx; fpt[1] = fy; fpt[2] = fz;
      float* o = newxyz + ((size_t)b*2048 + j)*3;
      o[0] = fx; o[1] = fy; o[2] = fz;
    }
    __syncthreads();
    float fx = fpt[0], fy = fpt[1], fz = fpt[2];
    float best = -1.0f; int bi = 0;
    #pragma unroll
    for (int i = 0; i < FPS_PPT; ++i) {
      float dx = lx[i]-fx, dy = ly[i]-fy, dz = lz[i]-fz;
      float d = dx*dx + dy*dy + dz*dz;
      ld[i] = fminf(ld[i], d);
      if (ld[i] > best) { best = ld[i]; bi = t + i*FPS_T; }
    }
    // intra-wave argmax (wave32 shuffles, no barriers)
    #pragma unroll
    for (int d = 16; d > 0; d >>= 1) {
      float ov = __shfl_down(best, d, 32);
      int   oi = __shfl_down(bi,   d, 32);
      if (ov > best || (ov == best && oi < bi)) { best = ov; bi = oi; }
    }
    if (lane == 0) { sv[wid] = best; si[wid] = bi; }
    __syncthreads();
    if (wid == 0) {
      float v2 = sv[lane]; int i2 = si[lane];
      #pragma unroll
      for (int d = 16; d > 0; d >>= 1) {
        float ov = __shfl_down(v2, d, 32);
        int   oi = __shfl_down(i2, d, 32);
        if (ov > v2 || (ov == v2 && oi < i2)) { v2 = ov; i2 = oi; }
      }
      if (lane == 0) sfar = i2;
    }
    __syncthreads();
  }
}

// ======================= Ball query (all 3 radii, one scan) =======================
__global__ void ball_query_kernel(const float* __restrict__ xyz,
                                  const float* __restrict__ newxyz,
                                  int* __restrict__ idx0, int* __restrict__ idx1,
                                  int* __restrict__ idx2) {
  int q = blockIdx.x * blockDim.x + threadIdx.x;   // (b*S + s)
  if (q >= B_ * S_) return;
  int b = q / S_;
  const float* nq = newxyz + (size_t)q * 3;
  float qx = nq[0], qy = nq[1], qz = nq[2];
  const float* px = xyz + (size_t)b * N_ * 3;
  int c0 = 0, c1 = 0, c2 = 0;
  int f0 = N_-1, f1 = N_-1, f2 = N_-1;
  int* o0 = idx0 + (size_t)q * 16;
  int* o1 = idx1 + (size_t)q * 32;
  int* o2 = idx2 + (size_t)q * 64;
  for (int n = 0; n < N_; ++n) {
    float dx = px[n*3+0]-qx, dy = px[n*3+1]-qy, dz = px[n*3+2]-qz;
    float d2 = dx*dx + dy*dy + dz*dz;
    if (d2 < 0.16f) {
      if (c2 < 64) { if (c2 == 0) f2 = n; o2[c2++] = n; }
      if (d2 < 0.04f) {
        if (c1 < 32) { if (c1 == 0) f1 = n; o1[c1++] = n; }
        if (d2 < 0.01f) {
          if (c0 < 16) { if (c0 == 0) f0 = n; o0[c0++] = n; }
        }
      }
      if (c0 >= 16 && c1 >= 32 && c2 >= 64) break;
    }
  }
  for (int k = c0; k < 16; ++k) o0[k] = f0;
  for (int k = c1; k < 32; ++k) o1[k] = f1;
  for (int k = c2; k < 64; ++k) o2[k] = f2;
}

// ======================= features (B,C,N) f32 -> (B,N,C) f16 =======================
__global__ void feats_transpose_kernel(const float* __restrict__ feats,
                                       _Float16* __restrict__ ft) {
  int e = blockIdx.x * blockDim.x + threadIdx.x;   // (b*N + n)*64 + c, c fastest
  if (e >= B_ * N_ * C_) return;
  int c  = e % C_;
  int bn = e / C_;
  int n  = bn % N_;
  int b  = bn / N_;
  ft[e] = (_Float16)feats[((size_t)b*C_ + c)*N_ + n];
}

// ======================= Gather -> X0 f16 [P][96] =======================
// X0 channel order: 0..63 = features (128B contiguous copy), 64..66 = xyz-center,
// 67..95 = zero. W1 columns are permuted to match.
__global__ void gather_kernel(const float* __restrict__ xyz,
                              const _Float16* __restrict__ ft,
                              const float* __restrict__ newxyz,
                              const int* __restrict__ idx,
                              _Float16* __restrict__ X, int K, int P) {
  int p = blockIdx.x * blockDim.x + threadIdx.x;
  if (p >= P) return;
  int q = p / K;            // (b*S + s)
  int b = q / S_;
  int j = idx[p];
  _Float16* xo = X + (size_t)p * 96;
  const uint4* src = (const uint4*)(ft + ((size_t)b*N_ + j) * 64);  // 128B aligned
  uint4* dst = (uint4*)xo;                                          // 16B aligned
  #pragma unroll
  for (int w = 0; w < 8; ++w) dst[w] = src[w];
  const float* nq = newxyz + (size_t)q * 3;
  const float* pj = xyz + ((size_t)b*N_ + j) * 3;
  xo[64] = (_Float16)(pj[0] - nq[0]);
  xo[65] = (_Float16)(pj[1] - nq[1]);
  xo[66] = (_Float16)(pj[2] - nq[2]);
  #pragma unroll
  for (int c = 67; c < 96; ++c) xo[c] = (_Float16)0.0f;
}

// ======================= weight conversion =======================
// W1: permute columns (feat first, then xyz), pad 67->96.
__global__ void convert_w1_kernel(const float* __restrict__ w1, _Float16* __restrict__ dst) {
  int e = blockIdx.x * blockDim.x + threadIdx.x;
  if (e >= 3*64*96) return;
  int c  = e % 96;
  int o  = (e / 96) % 64;
  int si = e / (96 * 64);
  int srcc = (c < 64) ? (c + 3) : ((c < 67) ? (c - 64) : -1);
  float v = (srcc >= 0) ? w1[((size_t)si*64 + o)*67 + srcc] : 0.0f;
  dst[e] = (_Float16)v;
}

__global__ void convert_pad_kernel(const float* __restrict__ src, _Float16* __restrict__ dst,
                                   int O, int Cin, int Cpad, int total) {
  int e = blockIdx.x * blockDim.x + threadIdx.x;
  if (e >= total) return;
  int c  = e % Cpad;
  int oo = (e / Cpad) % O;
  int si = e / (Cpad * O);
  float v = (c < Cin) ? src[((size_t)si*O + oo)*Cin + c] : 0.0f;
  dst[e] = (_Float16)v;
}

// ======================= WMMA GEMM: Y[P][M] = W[M][KC] * X[P][KC]^T =======================
// One wave per 16x(16*NT) strip: weight fragments loaded once (all k-steps in
// registers), NT B-tiles streamed through them. KT*NT unrolled v_wmma per wave.
template<int M, int KC, int NT>
__global__ void __launch_bounds__(32)
wmma_gemm_kernel(const _Float16* __restrict__ W, const _Float16* __restrict__ X,
                 float* __restrict__ Y, int P) {
  constexpr int KT = KC / 32;
  int lane = threadIdx.x;
  int half = lane >> 4;
  int lw   = lane & 15;
  int n0    = blockIdx.x * (16 * NT);
  int mbase = blockIdx.y * 16;
  union U { v16h h; unsigned int u[8]; };
  // A 16x32 f16 fragment: VGPR v holds K = {base,base+1}, base = (v<4?0:16)+(half?8:0)+2*(v&3)
  U a[KT];
  const _Float16* wrow = W + (size_t)(mbase + lw) * KC;
  #pragma unroll
  for (int kt = 0; kt < KT; ++kt) {
    #pragma unroll
    for (int v = 0; v < 8; ++v) {
      int ka = ((v & 4) << 2) + (half << 3) + ((v & 3) << 1);
      a[kt].u[v] = *(const unsigned int*)(wrow + kt*32 + ka);
    }
  }
  #pragma unroll
  for (int nt = 0; nt < NT; ++nt) {
    int nbase = n0 + nt * 16;
    const _Float16* xcol = X + (size_t)(nbase + lw) * KC;
    v8f acc = {0.f,0.f,0.f,0.f,0.f,0.f,0.f,0.f};
    #pragma unroll
    for (int kt = 0; kt < KT; ++kt) {
      U bf;
      #pragma unroll
      for (int v = 0; v < 8; ++v) {
        // B 32x16 f16 fragment: VGPR v holds K = {base,base+1}, base = (v<4?0:16)+(half?8:0)+2*(v&3)
        int kb = ((v & 4) << 2) + (half << 3) + ((v & 3) << 1);
        bf.u[v] = *(const unsigned int*)(xcol + kt*32 + kb);
      }
      acc = __builtin_amdgcn_wmma_f32_16x16x32_f16(
          false, a[kt].h, false, bf.h, (short)0, acc, false, false);
    }
    // D: lane holds column n = lane%16, rows m = mbase + half*8 + r
    int p  = nbase + lw;
    int m0 = mbase + (half << 3);
    float* dst = Y + (size_t)p * M + m0;
    *(float4*)(dst)     = make_float4(acc[0], acc[1], acc[2], acc[3]);
    *(float4*)(dst + 4) = make_float4(acc[4], acc[5], acc[6], acc[7]);
  }
}

// ======================= BN statistics =======================
__global__ void zero_stats_kernel(float* sums, float* sumsq) {
  int t = threadIdx.x;
  if (t < 128) { sums[t] = 0.f; sumsq[t] = 0.f; }
}

// blockDim = 384 (multiple of 64/96/128): each thread's channel is constant
// along the grid stride; LDS-reduced, one global f32 atomic per channel/block.
__global__ void stats_kernel(const float* __restrict__ Y, float* sums, float* sumsq,
                             int Cout, long long total) {
  __shared__ float ls[128], lq[128];
  int t = threadIdx.x;
  if (t < Cout) { ls[t] = 0.f; lq[t] = 0.f; }
  __syncthreads();
  long long gt = (long long)blockIdx.x * blockDim.x + t;
  long long T  = (long long)gridDim.x * blockDim.x;
  int c = (int)(gt % Cout);
  float s = 0.f, q = 0.f;
  for (long long e = gt; e < total; e += T) {
    float v = Y[e];
    s += v; q += v * v;
  }
  atomicAdd(&ls[c], s);
  atomicAdd(&lq[c], q);
  __syncthreads();
  if (t < Cout) { atomicAdd(&sums[t], ls[t]); atomicAdd(&sumsq[t], lq[t]); }
}

// ======================= BN + ReLU -> next-layer f16 input =======================
__global__ void bn_relu_kernel(const float* __restrict__ Y, _Float16* __restrict__ Xn,
                               const float* __restrict__ g, const float* __restrict__ bb,
                               const float* __restrict__ sums, const float* __restrict__ sumsq,
                               int Cout, int P, long long total) {
  long long gt = (long long)blockIdx.x * blockDim.x + threadIdx.x;
  long long T  = (long long)gridDim.x * blockDim.x;
  float invP = 1.0f / (float)P;
  for (long long e = gt; e < total; e += T) {
    int c = (int)(e % Cout);
    float mu  = sums[c] * invP;
    float var = sumsq[c] * invP - mu * mu;
    float rs  = rsqrtf(var + 1e-5f);
    float v = (Y[e] - mu) * rs * g[c] + bb[c];
    Xn[e] = (_Float16)fmaxf(v, 0.0f);
  }
}

// ======================= Final BN + ReLU + max over K -> output =======================
// channel-fastest mapping: K-deep reads of Y[.][128] are coalesced.
__global__ void bn_relu_max_kernel(const float* __restrict__ Y,
                                   const float* __restrict__ g, const float* __restrict__ bb,
                                   const float* __restrict__ sums, const float* __restrict__ sumsq,
                                   float* __restrict__ out, int K, int P, int scaleOff) {
  int gt = blockIdx.x * blockDim.x + threadIdx.x;
  if (gt >= B_ * 128 * S_) return;
  int c = gt % 128;
  int s = (gt / 128) % S_;
  int b = gt / (128 * S_);
  float invP = 1.0f / (float)P;
  float mu  = sums[c] * invP;
  float var = sumsq[c] * invP - mu * mu;
  float rs  = rsqrtf(var + 1e-5f);
  float gc = g[c], bc = bb[c];
  float m = 0.0f;  // post-ReLU values are >= 0
  long long base = (long long)(b*S_ + s) * K;
  for (int k = 0; k < K; ++k) {
    float v = Y[(base + k) * 128 + c];
    v = fmaxf((v - mu) * rs * gc + bc, 0.0f);
    m = fmaxf(m, v);
  }
  out[((size_t)b * 384 + scaleOff + c) * S_ + s] = m;
}

// ======================= Host orchestration =======================
extern "C" void kernel_launch(void* const* d_in, const int* in_sizes, int n_in,
                              void* d_out, int out_size, void* d_ws, size_t ws_size,
                              hipStream_t stream) {
  const float* xyz   = (const float*)d_in[0];
  const float* feats = (const float*)d_in[1];
  const float* w1 = (const float*)d_in[2];
  const float* g1 = (const float*)d_in[3];
  const float* b1 = (const float*)d_in[4];
  const float* w2 = (const float*)d_in[5];
  const float* g2 = (const float*)d_in[6];
  const float* b2 = (const float*)d_in[7];
  const float* w3 = (const float*)d_in[8];
  const float* g3 = (const float*)d_in[9];
  const float* b3 = (const float*)d_in[10];
  float* out = (float*)d_out;
  char*  ws  = (char*)d_ws;

  size_t off = 0;
  auto alloc = [&](size_t bytes) { size_t o = off; off = (off + bytes + 255) & ~(size_t)255; return o; };
  float*     sums  = (float*)    (ws + alloc(128 * 4));
  float*     sumsq = (float*)    (ws + alloc(128 * 4));
  _Float16*  W1h   = (_Float16*) (ws + alloc((size_t)3*64*96*2));
  _Float16*  W2h   = (_Float16*) (ws + alloc((size_t)3*96*64*2));
  _Float16*  W3h   = (_Float16*) (ws + alloc((size_t)3*128*96*2));
  _Float16*  FTh   = (_Float16*) (ws + alloc((size_t)B_*N_*C_*2));      // 4 MB
  int*       idx0  = (int*)      (ws + alloc((size_t)B_*S_*16*4));
  int*       idx1  = (int*)      (ws + alloc((size_t)B_*S_*32*4));
  int*       idx2  = (int*)      (ws + alloc((size_t)B_*S_*64*4));
  _Float16*  Xb    = (_Float16*) (ws + alloc((size_t)B_*S_*64*96*2));   // ~96 MB
  float*     Yb    = (float*)    (ws + alloc((size_t)B_*S_*64*128*4));  // ~256 MB

  float* newxyz  = out;                         // (B, 2048, 3)
  float* outFeat = out + (size_t)B_ * 2048 * 3; // (B, 384, 2048)

  fps_kernel<<<B_, FPS_T, 0, stream>>>(xyz, newxyz);

  { int tot = 3*64*96;  convert_w1_kernel<<<(tot+255)/256, 256, 0, stream>>>(w1, W1h); }
  { int tot = 3*96*64;  convert_pad_kernel<<<(tot+255)/256, 256, 0, stream>>>(w2, W2h, 96, 64, 64, tot); }
  { int tot = 3*128*96; convert_pad_kernel<<<(tot+255)/256, 256, 0, stream>>>(w3, W3h, 128, 96, 96, tot); }
  { int tot = B_*N_*C_; feats_transpose_kernel<<<(tot+255)/256, 256, 0, stream>>>(feats, FTh); }

  ball_query_kernel<<<(B_*S_ + 255)/256, 256, 0, stream>>>(xyz, newxyz, idx0, idx1, idx2);

  const int Ks[3] = {16, 32, 64};
  int* idxs[3] = {idx0, idx1, idx2};

  for (int i = 0; i < 3; ++i) {
    int K = Ks[i];
    int P = B_ * S_ * K;

    gather_kernel<<<(P + 255)/256, 256, 0, stream>>>(xyz, FTh, newxyz, idxs[i], Xb, K, P);

    // ---- Layer 1: 96(pad 67, permuted) -> 64 ----
    {
      dim3 grid(P/64, 64/16);
      wmma_gemm_kernel<64,96,4><<<grid, 32, 0, stream>>>(W1h + (size_t)i*64*96, Xb, Yb, P);
      zero_stats_kernel<<<1, 128, 0, stream>>>(sums, sumsq);
      long long tot = (long long)P * 64;
      stats_kernel<<<64, 384, 0, stream>>>(Yb, sums, sumsq, 64, tot);
      bn_relu_kernel<<<1024, 256, 0, stream>>>(Yb, Xb, g1 + i*64, b1 + i*64, sums, sumsq, 64, P, tot);
    }
    // ---- Layer 2: 64 -> 96 ----
    {
      dim3 grid(P/64, 96/16);
      wmma_gemm_kernel<96,64,4><<<grid, 32, 0, stream>>>(W2h + (size_t)i*96*64, Xb, Yb, P);
      zero_stats_kernel<<<1, 128, 0, stream>>>(sums, sumsq);
      long long tot = (long long)P * 96;
      stats_kernel<<<64, 384, 0, stream>>>(Yb, sums, sumsq, 96, tot);
      bn_relu_kernel<<<1024, 256, 0, stream>>>(Yb, Xb, g2 + i*96, b2 + i*96, sums, sumsq, 96, P, tot);
    }
    // ---- Layer 3: 96 -> 128, then max over K ----
    {
      dim3 grid(P/64, 128/16);
      wmma_gemm_kernel<128,96,4><<<grid, 32, 0, stream>>>(W3h + (size_t)i*128*96, Xb, Yb, P);
      zero_stats_kernel<<<1, 128, 0, stream>>>(sums, sumsq);
      long long tot = (long long)P * 128;
      stats_kernel<<<64, 384, 0, stream>>>(Yb, sums, sumsq, 128, tot);
      bn_relu_max_kernel<<<(B_*128*S_ + 255)/256, 256, 0, stream>>>(
          Yb, g3 + i*128, b3 + i*128, sums, sumsq, outFeat, K, P, i*128);
    }
  }
}
